// CrossAttention_10075993277015
// MI455X (gfx1250) — matrix operand order repaired and checked by reading, not compiled, not run on hardware
//
#include <hip/hip_runtime.h>
#include <hip/hip_bf16.h>
#include <math.h>

// ---------------------------------------------------------------------------
// CrossAttention for MI455X (gfx1250, wave32, WMMA bf16 16x16x32)
// b=2, sq=sk=2048, d=1024, h=16, hd=64
//
// Pipeline:
//   prep_kv : K fp32 [b,sk,d]  -> Kbf bf16 [b*h, sk, 64]   (head-major)
//             V fp32 [b,sk,d]  -> Vt  bf16 [b*h, 64, sk]   (LDS-tiled
//             transpose so both reads and writes are fully coalesced)
//   prep_wo : Wo fp32 [d,d]    -> Wobf bf16 [d,d]          (vectorized)
//   attn    : flash attention (kv tile = 64), base-2 softmax,
//             row-sum via P x ones WMMA, row-max via v_permlane16_b32,
//             out -> Xbf bf16 [b*sq, d]
//   proj    : out = Xbf @ Wobf^T + bo  (fp32 result)
// ---------------------------------------------------------------------------

typedef __attribute__((ext_vector_type(16))) __bf16 v16bf;
typedef __attribute__((ext_vector_type(8)))  __bf16 v8bf;
typedef __attribute__((ext_vector_type(8)))  float  v8f;

#define BATCH 2
#define SQ    2048
#define SK    2048
#define DIM   1024
#define NH    16
#define HD    64
#define KT    64                 // kv tile
#define WAVES_PER_BLOCK 8

#define NEG_BIG  (-3.0e38f)
#define NEG_THR  (-1.0e30f)
#define QSCALE   (0.18033688011112042f)   // (1/sqrt(64)) * log2(e)

__device__ __forceinline__ float exp2_fast(float x) {
#if __has_builtin(__builtin_amdgcn_exp2f)
  return __builtin_amdgcn_exp2f(x);
#else
  return exp2f(x);
#endif
}

// xor-butterfly lane permute within rows of 16 (VALU, no LDS traffic).
__device__ __forceinline__ float permx(float x, unsigned s0, unsigned s1,
                                       int xormask) {
#if __has_builtin(__builtin_amdgcn_permlane16)
  int xi = __builtin_bit_cast(int, x);
  int r  = __builtin_amdgcn_permlane16(xi, xi, s0, s1, false, false);
  (void)xormask;
  return __builtin_bit_cast(float, r);
#else
  (void)s0; (void)s1;
  return __shfl_xor(x, xormask, 32);
#endif
}

// max across the 16 lanes of a half-group (xor masks < 16 preserve halves).
__device__ __forceinline__ float rowmax16(float x) {
  x = fmaxf(x, permx(x, 0x67452301u, 0xEFCDAB89u, 1));  // xor 1
  x = fmaxf(x, permx(x, 0x54761032u, 0xDCFEBA98u, 2));  // xor 2
  x = fmaxf(x, permx(x, 0x32107654u, 0xBA98FEDCu, 4));  // xor 4
  x = fmaxf(x, permx(x, 0xFEDCBA98u, 0x76543210u, 8));  // xor 8
  return x;
}

// ---------------------------------------------------------------------------
// Prep: K -> head-major bf16 (direct, coalesced); V -> transposed bf16 via
// a 64x64 LDS tile so the Vt stores are contiguous 2B runs (coalesced),
// instead of 2B scatter at 4KB stride (64x write amplification).
// One block per (b*h, 64-key tile): 1024 blocks.
// ---------------------------------------------------------------------------
__global__ __launch_bounds__(256)
void prep_kv_kernel(const float* __restrict__ K, const float* __restrict__ V,
                    __bf16* __restrict__ Kbf, __bf16* __restrict__ Vt) {
  __shared__ __bf16 ldsV[64][66];   // pitch 66 spreads banks on column reads

  const int tid = threadIdx.x;
  const int bh  = blockIdx.x >> 5;          // b*NH + h
  const int s0  = (blockIdx.x & 31) * 64;   // key-tile origin
  const int b   = bh / NH;
  const int h   = bh % NH;

  const float* Kbase = K + ((size_t)b * SK + s0) * DIM + h * HD;
  const float* Vbase = V + ((size_t)b * SK + s0) * DIM + h * HD;
  __bf16* Kout = Kbf + ((size_t)bh * SK + s0) * HD;

#pragma unroll
  for (int i = 0; i < 16; ++i) {
    int idx = i * 256 + tid;
    int r = idx >> 6, c = idx & 63;          // consecutive tid -> consecutive c
    Kout[r * HD + c] = (__bf16)Kbase[(size_t)r * DIM + c];
    ldsV[r][c]       = (__bf16)Vbase[(size_t)r * DIM + c];
  }
  __syncthreads();
#pragma unroll
  for (int i = 0; i < 16; ++i) {
    int idx = i * 256 + tid;
    int c = idx >> 6, sl = idx & 63;         // consecutive tid -> consecutive sl
    Vt[((size_t)bh * HD + c) * SK + s0 + sl] = ldsV[sl][c];
  }
}

__global__ __launch_bounds__(256)
void prep_wo_kernel(const float* __restrict__ Wo, __bf16* __restrict__ Wobf) {
  size_t base = ((size_t)blockIdx.x * 256 + threadIdx.x) * 8;  // over DIM*DIM
  v8f x = *(const v8f*)(Wo + base);
  v8bf y;
#pragma unroll
  for (int i = 0; i < 8; ++i) y[i] = (__bf16)x[i];
  *(v8bf*)(Wobf + base) = y;
}

// ---------------------------------------------------------------------------
// Flash attention. Block = 8 waves = 8 q-tiles of the SAME head.
// ---------------------------------------------------------------------------
__global__ __launch_bounds__(256)
void attn_fa_kernel(const float* __restrict__ Q, const __bf16* __restrict__ Kbf,
                    const __bf16* __restrict__ Vt, const int* __restrict__ kmask,
                    const int* __restrict__ qmask, __bf16* __restrict__ Xbf) {
  __shared__ __align__(32) float lds_p[WAVES_PER_BLOCK][16 * KT];  // 32 KB

  const int lane = threadIdx.x & 31;
  const int warp = threadIdx.x >> 5;
  const int half = lane >> 4;     // 0 or 1
  const int l16  = lane & 15;

  // 512 blocks: 16 blocks per (b,h); 8 q-tiles per block.
  const int hidx = blockIdx.x >> 4;          // b*NH + h
  const int b    = hidx / NH;
  const int h    = hidx % NH;
  const int q0   = (((blockIdx.x & 15) * WAVES_PER_BLOCK) + warp) * 16;

  // ---- Q tile 16x64 -> two bf16 A operands; fold log2e/sqrt(hd) in.
  v16bf aQ[2];
  {
    const float* qrow = Q + ((size_t)b * SQ + q0 + l16) * DIM + h * HD;
#pragma unroll
    for (int c = 0; c < 2; ++c) {
      v8f lo = *(const v8f*)(qrow + c * 32 + half * 16);
      v8f hi = *(const v8f*)(qrow + c * 32 + half * 16 + 8);
#pragma unroll
      for (int i = 0; i < 8; ++i) {
        aQ[c][i]     = (__bf16)(lo[i] * QSCALE);
        aQ[c][8 + i] = (__bf16)(hi[i] * QSCALE);
      }
    }
  }

  // All-ones B operand for row-sum-by-WMMA.
  v16bf bOnes;
#pragma unroll
  for (int i = 0; i < 16; ++i) bOnes[i] = (__bf16)1.0f;

  v8f o[4] = {v8f{}, v8f{}, v8f{}, v8f{}};
  float m[8], l[8];
#pragma unroll
  for (int r = 0; r < 8; ++r) { m[r] = NEG_BIG; l[r] = 0.0f; }

  const __bf16* Kb  = Kbf + (size_t)hidx * SK * HD;
  const __bf16* Vtb = Vt  + (size_t)hidx * HD * SK;
  const int*    kmb = kmask + (size_t)b * SK;
  float* myp = lds_p[warp];

  for (int kv0 = 0; kv0 < SK; kv0 += KT) {
    if (kv0 + KT < SK) {
      __builtin_prefetch(Kb + (size_t)(kv0 + KT + lane * 2) * HD, 0, 1);
      __builtin_prefetch(Vtb + (size_t)(lane * 2) * SK + kv0 + KT, 0, 1);
    }

    // ---- S = Q*K^T for four 16-key groups (8 WMMAs), base-2 scaled.
    v8f s[4];
#pragma unroll
    for (int g = 0; g < 4; ++g) {
      v8f acc = {};
      const __bf16* krow = Kb + (size_t)(kv0 + g * 16 + l16) * HD + half * 16;
#pragma unroll
      for (int c = 0; c < 2; ++c) {
        v16bf bK = *(const v16bf*)(krow + c * 32);
        acc = __builtin_amdgcn_wmma_f32_16x16x32_bf16(
            false, aQ[c], false, bK, (short)0, acc, false, false);
      }
      s[g] = acc;
    }

    // ---- key mask: lane's C-layout column in group g is key kv0+g*16+l16.
#pragma unroll
    for (int g = 0; g < 4; ++g) {
      if (kmb[kv0 + g * 16 + l16] == 0) {
#pragma unroll
        for (int r = 0; r < 8; ++r) s[g][r] = NEG_BIG;
      }
    }

    // ---- online softmax (row r + 8*half). Masked entries underflow to 0;
    // fully-masked-so-far rows self-heal via alpha=0 on first valid tile.
#pragma unroll
    for (int r = 0; r < 8; ++r) {
      float t = fmaxf(fmaxf(s[0][r], s[1][r]), fmaxf(s[2][r], s[3][r]));
      float nm = fmaxf(m[r], rowmax16(t));
      float alpha = exp2_fast(m[r] - nm);
      m[r] = nm;
      l[r] *= alpha;
      o[0][r] *= alpha; o[1][r] *= alpha; o[2][r] *= alpha; o[3][r] *= alpha;
      s[0][r] = exp2_fast(s[0][r] - nm);
      s[1][r] = exp2_fast(s[1][r] - nm);
      s[2][r] = exp2_fast(s[2][r] - nm);
      s[3][r] = exp2_fast(s[3][r] - nm);
    }

    // ---- P: C-layout -> A-layout through per-wave LDS patch (16x64 f32).
#pragma unroll
    for (int g = 0; g < 4; ++g)
#pragma unroll
      for (int r = 0; r < 8; ++r)
        myp[(r + 8 * half) * KT + g * 16 + l16] = s[g][r];
    __syncthreads();
    v16bf aP[2];
#pragma unroll
    for (int c = 0; c < 2; ++c) {
      v8f plo = *(const v8f*)(myp + l16 * KT + c * 32 + half * 16);
      v8f phi = *(const v8f*)(myp + l16 * KT + c * 32 + half * 16 + 8);
#pragma unroll
      for (int i = 0; i < 8; ++i) {
        aP[c][i]     = (__bf16)plo[i];
        aP[c][8 + i] = (__bf16)phi[i];
      }
    }
    __syncthreads();

    // ---- row sums of P via WMMA against all-ones B (2 WMMAs).
    v8f ts = {};
    ts = __builtin_amdgcn_wmma_f32_16x16x32_bf16(
        false, aP[0], false, bOnes, (short)0, ts, false, false);
    ts = __builtin_amdgcn_wmma_f32_16x16x32_bf16(
        false, aP[1], false, bOnes, (short)0, ts, false, false);
#pragma unroll
    for (int r = 0; r < 8; ++r) l[r] += ts[r];

    // ---- O += P * V (transposed V: contiguous 32B loads; 8 WMMAs).
#pragma unroll
    for (int c = 0; c < 2; ++c)
#pragma unroll
      for (int g4 = 0; g4 < 4; ++g4) {
        v16bf bV = *(const v16bf*)(Vtb + (size_t)(g4 * 16 + l16) * SK +
                                   kv0 + c * 32 + half * 16);
        o[g4] = __builtin_amdgcn_wmma_f32_16x16x32_bf16(
            false, aP[c], false, bV, (short)0, o[g4], false, false);
      }
  }

  // ---- normalize, apply query mask / fully-masked-row zeroing, store bf16.
  const int* qmb = qmask + (size_t)b * SQ;
#pragma unroll
  for (int r = 0; r < 8; ++r) {
    int row = q0 + r + 8 * half;
    bool valid = (qmb[row] != 0) && (l[r] > 0.0f) && (m[r] > NEG_THR);
    float inv = valid ? (1.0f / l[r]) : 0.0f;
    __bf16* xrow = Xbf + ((size_t)b * SQ + row) * DIM + h * HD;
#pragma unroll
    for (int g4 = 0; g4 < 4; ++g4)
      xrow[g4 * 16 + l16] = (__bf16)(o[g4][r] * inv);
  }
}

// ---------------------------------------------------------------------------
// Projection: out = X @ Wo^T + bo  (bf16 WMMA GEMM, wave = 16x64 tile)
// ---------------------------------------------------------------------------
__global__ __launch_bounds__(256)
void proj_kernel(const __bf16* __restrict__ Xbf, const __bf16* __restrict__ Wobf,
                 const float* __restrict__ bo, float* __restrict__ out) {
  const int lane = threadIdx.x & 31;
  const int warp = threadIdx.x >> 5;
  const int half = lane >> 4;
  const int l16  = lane & 15;

  const int w  = blockIdx.x * WAVES_PER_BLOCK + warp;   // 0..4095
  const int mt = w / (DIM / 64);
  const int ng = w % (DIM / 64);
  const int m0 = mt * 16;
  const int n0 = ng * 64;

  v8f acc[4] = {v8f{}, v8f{}, v8f{}, v8f{}};
  const __bf16* xrow = Xbf + (size_t)(m0 + l16) * DIM;

  for (int kc = 0; kc < DIM; kc += 32) {
    v16bf aX;
    {
      v8bf lo = *(const v8bf*)(xrow + kc + half * 16);
      v8bf hi = *(const v8bf*)(xrow + kc + half * 16 + 8);
#pragma unroll
      for (int i = 0; i < 8; ++i) { aX[i] = lo[i]; aX[8 + i] = hi[i]; }
    }
#pragma unroll
    for (int g = 0; g < 4; ++g) {
      v16bf bW = *(const v16bf*)(Wobf + (size_t)(n0 + g * 16 + l16) * DIM +
                                 kc + half * 16);
      acc[g] = __builtin_amdgcn_wmma_f32_16x16x32_bf16(
          false, aX, false, bW, (short)0, acc[g], false, false);
    }
  }

#pragma unroll
  for (int g = 0; g < 4; ++g)
#pragma unroll
    for (int r = 0; r < 8; ++r) {
      int row = m0 + r + 8 * half;
      int col = n0 + g * 16 + l16;
      out[(size_t)row * DIM + col] = acc[g][r] + bo[col];
    }
}

// ---------------------------------------------------------------------------
extern "C" void kernel_launch(void* const* d_in, const int* in_sizes, int n_in,
                              void* d_out, int out_size, void* d_ws, size_t ws_size,
                              hipStream_t stream) {
  const float* Q     = (const float*)d_in[0];
  const float* K     = (const float*)d_in[1];
  const float* V     = (const float*)d_in[2];
  const int*   kmask = (const int*)d_in[3];
  const int*   qmask = (const int*)d_in[4];
  const float* Wo    = (const float*)d_in[5];
  const float* bo    = (const float*)d_in[6];
  float* out = (float*)d_out;

  // Workspace carve-up (26 MB total):
  char* ws = (char*)d_ws;
  __bf16* Kbf  = (__bf16*)(ws);                       // 8 MB  [b*h, sk, 64]
  __bf16* Vt   = (__bf16*)(ws + (8u << 20));          // 8 MB  [b*h, 64, sk]
  __bf16* Xbf  = (__bf16*)(ws + (16u << 20));         // 8 MB  [b*sq, d]
  __bf16* Wobf = (__bf16*)(ws + (24u << 20));         // 2 MB  [d, d]

  // 1024 blocks: one 64-key x 64-dim tile per (b*h, key-tile)
  prep_kv_kernel<<<1024, 256, 0, stream>>>(K, V, Kbf, Vt);
  // DIM*DIM / (256*8) = 512 blocks
  prep_wo_kernel<<<512, 256, 0, stream>>>(Wo, Wobf);
  // 512 blocks: 32 heads x 16 blocks, 8 q-tile waves per block
  attn_fa_kernel<<<512, 256, 0, stream>>>(Q, Kbf, Vt, kmask, qmask, Xbf);
  // 4096 output-tile waves / 8 = 512 blocks
  proj_kernel<<<512, 256, 0, stream>>>(Xbf, Wobf, bo, out);
}
